// CRFBiRNNModel_22265110463209
// MI455X (gfx1250) — matrix operand order repaired and checked by reading, not compiled, not run on hardware
//
#include <hip/hip_runtime.h>
#include <hip/hip_bf16.h>

#define K_    40
#define H_    16
#define D_    128
#define KK_   1600   // K*K
#define UROW_ 289    // 1 + 2H + 2D
#define MROW_ 145    // 1 + H + D
#define OPAD_ 48     // K padded to 3 WMMA N-tiles
#define SLABK_ 320   // K-slab staged in LDS (5 slabs x 10 WMMA chunks)
#define LOG2E_ 1.44269504088896f

typedef __attribute__((ext_vector_type(16))) _Float16 v16h;
typedef __attribute__((ext_vector_type(8)))  float    v8f;

#if defined(__has_builtin)
#if __has_builtin(__builtin_amdgcn_global_load_async_to_lds_b128) && \
    __has_builtin(__builtin_amdgcn_s_wait_asynccnt)
#define USE_ASYNC_COPY 1
#endif
#endif

#ifdef USE_ASYNC_COPY
// param type per hipcc diagnostic: int __attribute__((vector_size(16))) *
typedef __attribute__((__vector_size__(4 * sizeof(int)))) int v4i_;
typedef __attribute__((address_space(3))) v4i_ lv4i_;
#endif

// fast sigmoid: v_exp_f32 (exp2) + v_rcp_f32, no IEEE divide sequence
__device__ __forceinline__ float sigmoidf_(float x) {
  return __builtin_amdgcn_rcpf(1.0f + __builtin_amdgcn_exp2f(-LOG2E_ * x));
}

// ---------------------------------------------------------------- W = E[words]
__global__ void gather_w_kernel(const float* __restrict__ E,
                                const int* __restrict__ words,
                                float* __restrict__ W) {
  const int i = blockIdx.x;
  const int d = threadIdx.x;
  W[i * D_ + d] = E[(size_t)words[i] * D_ + d];
}

// ------------------------------------------------- forward RNN (serial scan)
__global__ __launch_bounds__(32) void fwd_rnn_kernel(const float* __restrict__ M,
                                                     const float* __restrict__ W,
                                                     float* __restrict__ hseq, int n) {
  __shared__ float hcur[H_];
  const int tid = threadIdx.x;
  if (tid < H_) hcur[tid] = 0.0f;
  __syncthreads();
  for (int t = 0; t < n; ++t) {
    float acc = 0.0f;
    if (tid < H_) {
      const float* mrow = M + tid * MROW_;
      acc = mrow[0];
      #pragma unroll
      for (int k = 0; k < H_; ++k) acc += mrow[1 + k] * hcur[k];
      const float* w = W + t * D_;
      for (int d = 0; d < D_; ++d) acc += mrow[1 + H_ + d] * w[d];
      acc = sigmoidf_(acc);
    }
    __syncthreads();
    if (tid < H_) { hcur[tid] = acc; hseq[t * H_ + tid] = acc; }
    __syncthreads();
  }
}

// ------------------------------------------------ backward RNN (serial scan)
__global__ __launch_bounds__(32) void bwd_rnn_kernel(const float* __restrict__ MP,
                                                     const float* __restrict__ W,
                                                     float* __restrict__ hpseq, int n) {
  __shared__ float hcur[H_];
  const int tid = threadIdx.x;
  if (tid < H_) hcur[tid] = 0.0f;
  __syncthreads();
  for (int t = n - 1; t >= 0; --t) {
    float acc = 0.0f;
    if (tid < H_) {
      const float* mrow = MP + tid * MROW_;
      acc = mrow[0];
      const float* w = W + t * D_;
      for (int d = 0; d < D_; ++d) acc += mrow[1 + d] * w[d];
      #pragma unroll
      for (int k = 0; k < H_; ++k) acc += mrow[1 + D_ + k] * hcur[k];
      acc = sigmoidf_(acc);
    }
    __syncthreads();
    if (tid < H_) { hcur[tid] = acc; hpseq[t * H_ + tid] = acc; }
    __syncthreads();
  }
}

// ------------------------------------- SA = UsA@T.T, TA = UtA@T.T, TB = UtB@T.T
__global__ void sat_kernel(const float* __restrict__ UA, const float* __restrict__ UB,
                           const float* __restrict__ T,
                           float* __restrict__ SA, float* __restrict__ TA,
                           float* __restrict__ TB) {
  const int k = blockIdx.x;        // 0..39
  const int which = blockIdx.y;    // 0:SA 1:TA 2:TB
  const int s = threadIdx.x;
  if (s >= K_) return;
  const float* urow;
  float* dst;
  if (which == 0)      { urow = UA + k * UROW_ + (1 + H_);        dst = SA; }
  else if (which == 1) { urow = UA + k * UROW_ + (1 + H_ + D_);   dst = TA; }
  else                 { urow = UB + k * UROW_ + (1 + H_);        dst = TB; }
  const float* trow = T + s * D_;
  float acc = 0.0f;
  for (int d = 0; d < D_; ++d) acc += urow[d] * trow[d];
  dst[k * K_ + s] = acc;
}

// ------------------------------------------------- preA / preB per-row kernel
__global__ void pre_kernel(const float* __restrict__ UA, const float* __restrict__ UB,
                           const float* __restrict__ hseq, const float* __restrict__ hpseq,
                           const float* __restrict__ W,
                           float* __restrict__ preA, float* __restrict__ preB, int n) {
  const int i = blockIdx.x;      // 0..n+1
  const int tid = threadIdx.x;   // 64 threads
  __shared__ float hrow[H_], hpArow[H_], hpBrow[H_], wrow[D_];
  if (tid < H_) {
    hrow[tid]   = (i >= 1 && i <= n) ? hseq[(i - 1) * H_ + tid] : 0.0f;
    hpArow[tid] = (i >= 3)           ? hpseq[(i - 3) * H_ + tid] : 0.0f;  // hpA = [0,0,hp[:n]]
    hpBrow[tid] = (i >= 2)           ? hpseq[(i - 2) * H_ + tid] : 0.0f;  // hpB = [0,hp[:n+1]]
  }
  for (int d = tid; d < D_; d += 64)
    wrow[d] = (i >= 1 && i <= n) ? W[(i - 1) * D_ + d] : 0.0f;
  __syncthreads();
  if (tid < K_) {
    const float* ua = UA + tid * UROW_;
    float a = ua[0];
    #pragma unroll
    for (int j = 0; j < H_; ++j) a += hrow[j] * ua[1 + j];
    #pragma unroll
    for (int j = 0; j < H_; ++j) a += hpArow[j] * ua[273 + j];
    preA[i * K_ + tid] = a;

    const float* ub = UB + tid * UROW_;
    float b = ub[0];
    #pragma unroll
    for (int j = 0; j < H_; ++j) b += hrow[j] * ub[1 + j];
    for (int d = 0; d < D_; ++d) b += wrow[d] * ub[1 + H_ + D_ + d];
    #pragma unroll
    for (int j = 0; j < H_; ++j) b += hpBrow[j] * ub[273 + j];
    preB[i * K_ + tid] = b;
  }
}

// ---------------------------------------- WlinA -> f16, padded to 48 rows (o)
__global__ void wlina_f16_kernel(const float* __restrict__ WlinA,
                                 _Float16* __restrict__ WlinAh) {
  const int idx = blockIdx.x * blockDim.x + threadIdx.x;
  if (idx >= OPAD_ * KK_) return;
  const int o = idx / KK_;
  const int c = idx - o * KK_;
  const float v = (o < K_) ? WlinA[o * KK_ + c] : 0.0f;
  WlinAh[idx] = (_Float16)v;
}

// ------------------------------- WBsumW[i,u] = sum_s WlinB[words[i], s, u]
__global__ void wbsum_kernel(const float* __restrict__ WlinB,
                             const int* __restrict__ words,
                             float* __restrict__ WBs) {
  const int i = blockIdx.x;
  const int u = threadIdx.x;
  if (u >= K_) return;
  const float* base = WlinB + (size_t)words[i] * KK_ + u;
  float acc = 0.0f;
  #pragma unroll
  for (int s = 0; s < K_; ++s) acc += base[s * K_];
  WBs[i * K_ + u] = acc;
}

// --------------- e[i,t] = sum_u sigmoid(preB[i,t] + TB[t,u]) * WBsumW[i,u]
__global__ void e_kernel(const float* __restrict__ preB, const float* __restrict__ TB,
                         const float* __restrict__ WBs, float* __restrict__ ev) {
  const int i = blockIdx.x;
  const int t = threadIdx.x;
  if (t >= K_) return;
  const float pb = preB[i * K_ + t];
  const float* tb = TB + t * K_;
  const float* wb = WBs + i * K_;
  float acc = 0.0f;
  #pragma unroll 8
  for (int u = 0; u < K_; ++u) acc += sigmoidf_(pb + tb[u]) * wb[u];
  ev[i * K_ + t] = acc;
}

// ===================== main WMMA kernel: logphiA = F @ WlinA.T (per r) =======
// 4 waves/block; wave w owns i-tile (blockIdx.x*4 + w): 16 rows(i) x 48 cols(o).
// ST[kk] = -log2e*(SA[r,s]+TA[r,t]) precomputed once per block in LDS.
// B (WlinA.T, f16) staged into LDS in 5 K-slabs via async global->LDS copies:
// steady-state loop is pure LDS + VALU/TRANS + WMMA, zero VMEM.
__global__ __launch_bounds__(128) void logphi_wmma_kernel(
    const float* __restrict__ preA, const float* __restrict__ SAall,
    const float* __restrict__ TAall, const _Float16* __restrict__ WlinAh,
    float* __restrict__ logphiA, int rows) {
  const int tid  = threadIdx.x;      // 0..127
  const int lane = tid & 31;
  const int wv   = tid >> 5;
  const int r    = blockIdx.y;

  __shared__ __attribute__((aligned(32))) float    STsh[KK_];           // 6.4 KB
  __shared__ __attribute__((aligned(32))) _Float16 Bsh[OPAD_ * SLABK_]; // 30.7 KB

  for (int j = tid; j < KK_; j += 128) {
    const int s = j / K_;
    const int t = j - s * K_;
    STsh[j] = -LOG2E_ * (SAall[r * K_ + s] + TAall[r * K_ + t]);
  }

  const int  itile = blockIdx.x * 4 + wv;
  const int  mrow  = lane & 15;
  const int  i     = itile * 16 + mrow;
  const bool valid = (i < rows);
  // invalid rows: exponent -> +inf -> exp2 -> inf -> rcp -> 0 (no divergence)
  const float pAs = valid ? (-LOG2E_ * preA[i * K_ + r]) : 1.0e30f;
  const int ka   = (lane >= 16) ? 16 : 0;   // A-frag K offset (contiguous k = 16h + s, matches B)
  const int kb   = (lane >= 16) ? 16 : 0;   // B-frag K offset (contiguous k = 16h + s, matches A)
  const int ocol = lane & 15;

  v8f c0 = {}, c1 = {}, c2 = {};
  for (int slab = 0; slab < KK_ / SLABK_; ++slab) {
    const int k0 = slab * SLABK_;
    // ---- stage B slab: 48 rows x 320 halfs (16B per op, 1920 ops/block) ----
#ifdef USE_ASYNC_COPY
    for (int idx = tid; idx < OPAD_ * SLABK_ / 8; idx += 128) {
      const int row = idx / (SLABK_ / 8);
      const int c8  = idx - row * (SLABK_ / 8);
      __builtin_amdgcn_global_load_async_to_lds_b128(
          (v4i_*)(WlinAh + (size_t)row * KK_ + k0 + c8 * 8),
          (lv4i_*)(Bsh + row * SLABK_ + c8 * 8), 0, 0);
    }
    __builtin_amdgcn_s_wait_asynccnt(0);
#else
    for (int idx = tid; idx < OPAD_ * SLABK_ / 8; idx += 128) {
      const int row = idx / (SLABK_ / 8);
      const int c8  = idx - row * (SLABK_ / 8);
      *(uint4*)(Bsh + row * SLABK_ + c8 * 8) =
          *(const uint4*)(WlinAh + (size_t)row * KK_ + k0 + c8 * 8);
    }
#endif
    __syncthreads();

    // ---- 10 WMMA chunks entirely from LDS ----
    for (int kk = 0; kk < SLABK_; kk += 32) {
      const int k = k0 + kk;
      const v8f st0 = *(const v8f*)(STsh + k + ka);        // hh 0..7  -> kk = k+ka+hh
      const v8f st1 = *(const v8f*)(STsh + k + ka + 8);    // hh 8..15 -> kk = k+ka+8+(hh-8)
      v16h a;
      #pragma unroll
      for (int hh = 0; hh < 8; ++hh)
        a[hh] = (_Float16)__builtin_amdgcn_rcpf(1.0f + __builtin_amdgcn_exp2f(pAs + st0[hh]));
      #pragma unroll
      for (int hh = 0; hh < 8; ++hh)
        a[8 + hh] = (_Float16)__builtin_amdgcn_rcpf(1.0f + __builtin_amdgcn_exp2f(pAs + st1[hh]));

      const _Float16* bp = Bsh + kk + kb;
      const v16h b0 = *(const v16h*)(bp + (ocol)      * SLABK_);
      const v16h b1 = *(const v16h*)(bp + (ocol + 16) * SLABK_);
      const v16h b2 = *(const v16h*)(bp + (ocol + 32) * SLABK_);
      c0 = __builtin_amdgcn_wmma_f32_16x16x32_f16(false, a, false, b0, (short)0, c0, false, false);
      c1 = __builtin_amdgcn_wmma_f32_16x16x32_f16(false, a, false, b1, (short)0, c1, false, false);
      c2 = __builtin_amdgcn_wmma_f32_16x16x32_f16(false, a, false, b2, (short)0, c2, false, false);
    }
    __syncthreads();
  }

  // Store D (f32 16x16 C/D layout): VGPR v -> row v + (lane>=16 ? 8 : 0).
  // logphiA rows are padded to gridDim.x*64, so row stores are unconditional.
  const int rbase = itile * 16 + ((lane >= 16) ? 8 : 0);
  #pragma unroll
  for (int v = 0; v < 8; ++v) {
    float* dst = logphiA + (size_t)(rbase + v) * KK_ + r * K_;
    dst[ocol] = c0[v];                              // o in [0,16)
    dst[ocol + 16] = c1[v];                         // o in [16,32)
    if (ocol + 32 < K_) dst[ocol + 32] = c2[v];     // o in [32,40)
  }
}

// ------------------------------------------ CRF forward scan + final logsumexp
__global__ __launch_bounds__(64) void crf_kernel(const float* __restrict__ logphiA,
                                                 const float* __restrict__ ev,
                                                 int n, float* __restrict__ out) {
  __shared__ float la[K_];
  __shared__ float red[64];
  const int tid = threadIdx.x;
  const float NEG = -1.0e30f;

  if (tid < K_) la[tid] = logphiA[0 * KK_ + 0 * K_ + tid] + ev[tid];   // BOS_T = 0
  __syncthreads();

  for (int i = 1; i < n; ++i) {
    float lanew = 0.0f;
    if (tid < K_) {
      const float* phi = logphiA + (size_t)i * KK_;   // [r*K + o]
      float m = NEG;
      #pragma unroll 8
      for (int r = 0; r < K_; ++r) m = fmaxf(m, la[r] + phi[r * K_ + tid]);
      float s = 0.0f;
      #pragma unroll 8
      for (int r = 0; r < K_; ++r) s += __expf(la[r] + phi[r * K_ + tid] - m);
      lanew = m + __logf(s) + ev[i * K_ + tid];
    }
    __syncthreads();
    if (tid < K_) la[tid] = lanew;
    __syncthreads();
  }

  const float v = (tid < K_) ? la[tid] + logphiA[(size_t)n * KK_ + tid * K_ + 1]  // EOS_T = 1
                             : NEG;
  red[tid] = v;
  __syncthreads();
  for (int off = 32; off > 0; off >>= 1) {
    if (tid < off) red[tid] = fmaxf(red[tid], red[tid + off]);
    __syncthreads();
  }
  const float m = red[0];
  __syncthreads();
  red[tid] = (tid < K_) ? __expf(v - m) : 0.0f;
  __syncthreads();
  for (int off = 32; off > 0; off >>= 1) {
    if (tid < off) red[tid] += red[tid + off];
    __syncthreads();
  }
  if (tid == 0) out[0] = m + __logf(red[0]);
}

// ============================================================================
extern "C" void kernel_launch(void* const* d_in, const int* in_sizes, int n_in,
                              void* d_out, int out_size, void* d_ws, size_t ws_size,
                              hipStream_t stream) {
  (void)n_in; (void)out_size; (void)ws_size;
  const float* E     = (const float*)d_in[0];
  const float* M     = (const float*)d_in[1];
  const float* MP    = (const float*)d_in[2];
  const float* T     = (const float*)d_in[3];
  const float* UA    = (const float*)d_in[4];
  const float* UB    = (const float*)d_in[5];
  const float* WlinA = (const float*)d_in[6];
  const float* WlinB = (const float*)d_in[7];
  const int*   words = (const int*)d_in[8];
  const int n      = in_sizes[8];
  const int rows   = n + 2;
  const int itiles = (rows + 15) / 16;
  const int gx     = (itiles + 3) / 4;
  const int padrows = gx * 64;            // grid covers gx*4 tiles of 16 rows

  float* ws = (float*)d_ws;
  size_t off = 0;
  float* W       = ws + off; off += (size_t)n * D_;
  float* hseq    = ws + off; off += (size_t)n * H_;
  float* hpseq   = ws + off; off += (size_t)n * H_;
  float* preA    = ws + off; off += (size_t)rows * K_;
  float* preB    = ws + off; off += (size_t)rows * K_;
  float* SA      = ws + off; off += KK_;
  float* TA      = ws + off; off += KK_;
  float* TB      = ws + off; off += KK_;
  float* WBs     = ws + off; off += (size_t)n * K_;
  float* ev      = ws + off; off += (size_t)n * K_;
  float* logphiA = ws + off; off += (size_t)padrows * KK_;
  off = (off + 15) & ~(size_t)15;                 // 64B align for v16h loads
  _Float16* WlinAh = (_Float16*)(ws + off);

  gather_w_kernel<<<n, D_, 0, stream>>>(E, words, W);
  fwd_rnn_kernel<<<1, 32, 0, stream>>>(M, W, hseq, n);
  bwd_rnn_kernel<<<1, 32, 0, stream>>>(MP, W, hpseq, n);
  sat_kernel<<<dim3(K_, 3), 64, 0, stream>>>(UA, UB, T, SA, TA, TB);
  pre_kernel<<<rows, 64, 0, stream>>>(UA, UB, hseq, hpseq, W, preA, preB, n);
  wlina_f16_kernel<<<(OPAD_ * KK_ + 255) / 256, 256, 0, stream>>>(WlinA, WlinAh);
  wbsum_kernel<<<n, 64, 0, stream>>>(WlinB, words, WBs);
  e_kernel<<<n, 64, 0, stream>>>(preB, TB, WBs, ev);

  logphi_wmma_kernel<<<dim3(gx, K_), 128, 0, stream>>>(preA, SA, TA, WlinAh, logphiA, rows);

  crf_kernel<<<1, 64, 0, stream>>>(logphiA, ev, n, (float*)d_out);
}
